// BipartiteEdgeWithAngleFeatures_15607911153859
// MI455X (gfx1250) — matrix operand order repaired and checked by reading, not compiled, not run on hardware
//
#include <hip/hip_runtime.h>

typedef __attribute__((ext_vector_type(16))) _Float16 v16h;
typedef __attribute__((ext_vector_type(8)))  float    v8f;

#define NUM_RBF     50
#define HIDDEN      128
#define KPAD        64
#define CUT_HI_F    10.0f
#define ALPHA_F     0.5f              // 5.0 / (CUT_HI - CUT_LO)
#define PI_OVER_CUT 0.31415926535f    // pi / CUT_HI

// K index held by A-operand half 't' of a lane in half-wave 'g'
// (CDNA5 16-bit A 16x32 layout: VGPR j<4 -> K=8g+2j(+1), j>=4 -> K=16+8g+2(j-4)(+1))
__device__ __forceinline__ int a_k_of(int g, int t) {
    int j = t >> 1;
    int base = (j < 4) ? (8 * g + 2 * j) : (16 + 8 * g + 2 * (j - 4));
    return base + (t & 1);
}

// ---------------------------------------------------------------------------
// Hot kernel: each block owns exactly 128 in-range edges. NO bounds checks.
// Bias is folded into the GEMM via K-row 50 (A=1.0, B=bias[n]).
// ---------------------------------------------------------------------------
__global__ __launch_bounds__(256) void edge_rbf_wmma_kernel(
    const long long* __restrict__ ei_src,       // edge_index row 0
    const long long* __restrict__ ei_dst,       // edge_index row 1
    const float*     __restrict__ node_pos,     // [N_NODES, 3]
    const float*     __restrict__ group_pos,    // [N_GROUPS, 3]
    const float*     __restrict__ means,        // [50]
    const float*     __restrict__ betas,        // [50]
    const float*     __restrict__ W,            // [50, 128]
    const float*     __restrict__ bias,         // [128]
    long long*       __restrict__ idx0_out,     // idx_out row 0
    long long*       __restrict__ idx1_out,     // idx_out row 1
    float*           __restrict__ w_out,        // [E]
    float*           __restrict__ attr_out,     // [E, 128]
    float*           __restrict__ vecn_out)     // [E, 3]
{
    __shared__ __align__(32) _Float16 sWt[HIDDEN * KPAD]; // W^T, f16; k=50 row = bias
    __shared__ float sMean[KPAD];
    __shared__ float sBeta[KPAD];

    const int tid = threadIdx.x;

    // Stage W transposed (n-major, contiguous K) as f16.
    // k < 50: W[k][n]; k == 50: bias[n]; k > 50: 0.
    for (int i = tid; i < HIDDEN * KPAD; i += 256) {
        const int n = i >> 6;
        const int k = i & 63;
        float v = 0.0f;
        if (k < NUM_RBF)       v = W[k * HIDDEN + n];
        else if (k == NUM_RBF) v = bias[n];
        sWt[i] = (_Float16)v;
    }
    if (tid < KPAD) {
        sMean[tid] = (tid < NUM_RBF) ? means[tid] : 0.0f;
        sBeta[tid] = (tid < NUM_RBF) ? betas[tid] : 0.0f;
    }
    __syncthreads();

    const int wave = tid >> 5;
    const int lane = tid & 31;
    const int g    = lane >> 4;   // half-wave
    const int m    = lane & 15;   // edge-in-wave / N-in-tile
    const int rowW = wave * 16;   // first local edge of this wave (0..112)

    const long long blkEdge0 = (long long)blockIdx.x * 128;
    const int le = rowW + m;      // local edge 0..127 (for this lane's half)

    // ---- per-edge geometry (all 32 lanes compute for edge (lane&15); gathers are L2-resident)
    const long long src = ei_src[blkEdge0 + le];
    const long long dst = ei_dst[blkEdge0 + le];
    const float px = node_pos[src * 3 + 0] - group_pos[dst * 3 + 0];
    const float py = node_pos[src * 3 + 1] - group_pos[dst * 3 + 1];
    const float pz = node_pos[src * 3 + 2] - group_pos[dst * 3 + 2];
    const float d   = sqrtf(px * px + py * py + pz * pz);
    const float inv = 1.0f / d;

    if (g == 0) {
        float*     wb = w_out    + blkEdge0;
        float*     vb = vecn_out + blkEdge0 * 3;
        long long* i0 = idx0_out + blkEdge0;
        long long* i1 = idx1_out + blkEdge0;
        __builtin_nontemporal_store(d,        wb + le);
        __builtin_nontemporal_store(px * inv, vb + le * 3 + 0);
        __builtin_nontemporal_store(py * inv, vb + le * 3 + 1);
        __builtin_nontemporal_store(pz * inv, vb + le * 3 + 2);
        __builtin_nontemporal_store(src, i0 + le);
        __builtin_nontemporal_store(dst, i1 + le);
    }

    // ---- RBF features straight into WMMA A-operand registers (f16, K padded to 64)
    // Slot k==50 carries 1.0 so the bias row of W^T lands in the accumulator.
    const float x   = __expf(-ALPHA_F * d);                               // exp(alpha*(lo - d))
    const float cut = (d < CUT_HI_F) ? 0.5f * (__cosf(d * PI_OVER_CUT) + 1.0f) : 0.0f;

    v16h a0, a1;
    #pragma unroll
    for (int t = 0; t < 16; ++t) {
        const int k0 = 16 * g + t;
        const int k1 = k0 + 32;
        const float u0 = x - sMean[k0];
        const float u1 = x - sMean[k1];
        const float r0 = (k0 < NUM_RBF) ? cut * __expf(-sBeta[k0] * u0 * u0) : 0.0f;
        float r1;
        if (k1 < NUM_RBF)       r1 = cut * __expf(-sBeta[k1] * u1 * u1);
        else if (k1 == NUM_RBF) r1 = 1.0f;                 // bias row
        else                    r1 = 0.0f;
        a0[t] = (_Float16)r0;
        a1[t] = (_Float16)r1;
    }

    // ---- edge_attr = rbf @ W + b : 8 N-tiles x 2 WMMAs (K=0..31, 32..63)
    float* attrBlk = attr_out + (size_t)blkEdge0 * HIDDEN;  // block-uniform base
    const int offRow = (rowW + 8 * g) * HIDDEN + m;         // per-lane, tile-invariant part

    #pragma unroll
    for (int nb = 0; nb < 8; ++nb) {
        v8f c = {};                                         // inline-0 SRC2 on first WMMA
        // B operand: lane holds column N=m, K half selected by g; 16 contiguous f16 from LDS
        const v16h b0 = *(const v16h*)&sWt[(nb * 16 + m) * KPAD + 16 * g];
        const v16h b1 = *(const v16h*)&sWt[(nb * 16 + m) * KPAD + 32 + 16 * g];
        c = __builtin_amdgcn_wmma_f32_16x16x32_f16(false, a0, false, b0, (short)0, c, false, false);
        c = __builtin_amdgcn_wmma_f32_16x16x32_f16(false, a1, false, b1, (short)0, c, false, false);

        // C layout: M = 8*g + r, N = m ; shared addr VGPR + imm offsets, NT
        const int off0 = offRow + nb * 16;
        #pragma unroll
        for (int r = 0; r < 8; ++r)
            __builtin_nontemporal_store(c[r], attrBlk + off0 + r * HIDDEN);
    }
}

// ---------------------------------------------------------------------------
// Tail kernel: scalar path for the (at most 127) edges past the last full block.
// ---------------------------------------------------------------------------
__global__ __launch_bounds__(128) void edge_rbf_tail_kernel(
    const long long* __restrict__ ei_src,
    const long long* __restrict__ ei_dst,
    const float*     __restrict__ node_pos,
    const float*     __restrict__ group_pos,
    const float*     __restrict__ means,
    const float*     __restrict__ betas,
    const float*     __restrict__ W,
    const float*     __restrict__ bias,
    long long*       __restrict__ idx0_out,
    long long*       __restrict__ idx1_out,
    float*           __restrict__ w_out,
    float*           __restrict__ attr_out,
    float*           __restrict__ vecn_out,
    long long eStart, long long E)
{
    const long long e = eStart + threadIdx.x;
    if (e >= E) return;

    const long long src = ei_src[e];
    const long long dst = ei_dst[e];
    const float px = node_pos[src * 3 + 0] - group_pos[dst * 3 + 0];
    const float py = node_pos[src * 3 + 1] - group_pos[dst * 3 + 1];
    const float pz = node_pos[src * 3 + 2] - group_pos[dst * 3 + 2];
    const float d   = sqrtf(px * px + py * py + pz * pz);
    const float inv = 1.0f / d;

    w_out[e] = d;
    vecn_out[e * 3 + 0] = px * inv;
    vecn_out[e * 3 + 1] = py * inv;
    vecn_out[e * 3 + 2] = pz * inv;
    idx0_out[e] = src;
    idx1_out[e] = dst;

    const float x   = __expf(-ALPHA_F * d);
    const float cut = (d < CUT_HI_F) ? 0.5f * (__cosf(d * PI_OVER_CUT) + 1.0f) : 0.0f;

    float rbf[NUM_RBF];
    for (int k = 0; k < NUM_RBF; ++k) {
        const float u = x - means[k];
        rbf[k] = cut * __expf(-betas[k] * u * u);
    }
    float* row = attr_out + (size_t)e * HIDDEN;
    for (int n = 0; n < HIDDEN; ++n) {
        float acc = bias[n];
        for (int k = 0; k < NUM_RBF; ++k) acc += rbf[k] * W[k * HIDDEN + n];
        row[n] = acc;
    }
}

extern "C" void kernel_launch(void* const* d_in, const int* in_sizes, int n_in,
                              void* d_out, int out_size, void* d_ws, size_t ws_size,
                              hipStream_t stream) {
    const long long* edge_index = (const long long*)d_in[0];
    const float*     node_pos   = (const float*)d_in[1];
    const float*     group_pos  = (const float*)d_in[2];
    const float*     means      = (const float*)d_in[3];
    const float*     betas      = (const float*)d_in[4];
    const float*     W          = (const float*)d_in[5];
    const float*     b          = (const float*)d_in[6];

    const long long E = (long long)in_sizes[0] / 2;
    if (E <= 0) return;

    // Outputs concatenated flat in return order:
    // [edge_index int64 2E][edge_weight f32 E][edge_attr f32 E*128][edge_vec_normalized f32 E*3]
    char* o = (char*)d_out;
    long long* idx_out  = (long long*)o;
    float*     w_out    = (float*)(o + (size_t)2 * (size_t)E * sizeof(long long));
    float*     attr_out = w_out + E;
    float*     vecn_out = attr_out + (size_t)E * HIDDEN;

    const long long* ei_src = edge_index;
    const long long* ei_dst = edge_index + E;
    long long*       idx0   = idx_out;
    long long*       idx1   = idx_out + E;

    const long long fullBlocks = E / 128;       // 12500 for the reference shapes
    if (fullBlocks > 0) {
        edge_rbf_wmma_kernel<<<(unsigned)fullBlocks, 256, 0, stream>>>(
            ei_src, ei_dst, node_pos, group_pos, means, betas, W, b,
            idx0, idx1, w_out, attr_out, vecn_out);
    }
    const long long tail = E - fullBlocks * 128;
    if (tail > 0) {
        edge_rbf_tail_kernel<<<1, 128, 0, stream>>>(
            ei_src, ei_dst, node_pos, group_pos, means, betas, W, b,
            idx0, idx1, w_out, attr_out, vecn_out, fullBlocks * 128, E);
    }
}
